// DoubleNetBoth_9758165696787
// MI455X (gfx1250) — matrix operand order repaired and checked by reading, not compiled, not run on hardware
//
#include <hip/hip_runtime.h>
#include <math.h>

// ---------------- problem constants (match reference) ----------------
#define NNODE 50000
#define NEDGE 400000
#define NGRF  2048
#define NFD   128
#define EFD   16
#define FD    144      // NF + EF
#define EMBD  64
#define NHD   4
#define ZD    256      // H * EMB

enum { ACT_NONE = 0, ACT_RELU = 1, ACT_LEAKY01 = 2 };

typedef __bf16 bf16_t;
typedef bf16_t v16bf __attribute__((ext_vector_type(16)));
typedef bf16_t v8bf  __attribute__((ext_vector_type(8)));
typedef float  v8f   __attribute__((ext_vector_type(8)));

// ---------------- device helpers ----------------
__device__ __forceinline__ bf16_t f2bf(float f) {
  unsigned int u = __builtin_bit_cast(unsigned int, f);
  u += 0x7FFFu + ((u >> 16) & 1u);           // round-to-nearest-even
  unsigned short h = (unsigned short)(u >> 16);
  return __builtin_bit_cast(bf16_t, h);
}

__device__ __forceinline__ void atomicMaxF(float* addr, float val) {
  // sign-split trick: lowers to global_atomic_max_i32 / global_atomic_min_u32
  if (val >= 0.0f) atomicMax((int*)addr, __float_as_int(val));
  else             atomicMin((unsigned int*)addr, __float_as_uint(val));
}

__device__ __forceinline__ float leakyf(float x, float s) { return x > 0.0f ? x : s * x; }

// ---------------- generic fill ----------------
__global__ void k_fill(float* p, long long n, float v) {
  long long i = (long long)blockIdx.x * blockDim.x + threadIdx.x;
  if (i < n) p[i] = v;
}

// ---------------- A[M,K] f32 -> Abf[M,Kpad] bf16 (zero-padded K) ----------------
__global__ void k_cvt_abf(const float* __restrict__ A, bf16_t* __restrict__ Abf,
                          long long M, int K, int Kpad) {
  long long idx = (long long)blockIdx.x * blockDim.x + threadIdx.x;
  if (idx >= M * Kpad) return;
  int k = (int)(idx % Kpad);
  long long m = idx / Kpad;
  Abf[idx] = (k < K) ? f2bf(A[m * K + k]) : f2bf(0.0f);
}

// ---------------- pack W[K,Nc] (f32, row-major) into per-lane WMMA B fragments ----------------
// Wp layout: [kt][nt][lane 0..31][e 0..15] bf16, where K(lane,e) = 8*(lane>>4)+e (e<8) / 16+8*(lane>>4)+(e-8) (e>=8), N(lane)=lane&15
__global__ void k_pack_w(const float* __restrict__ W, bf16_t* __restrict__ Wp,
                         int K, int Nc, int numNT, int total) {
  int idx = blockIdx.x * blockDim.x + threadIdx.x;
  if (idx >= total) return;
  int e    = idx & 15;
  int lane = (idx >> 4) & 31;
  int t    = idx >> 9;
  int nt   = t % numNT;
  int kt   = t / numNT;
  int k = kt * 32 + (e < 8 ? (lane >> 4) * 8 + e : 16 + (lane >> 4) * 8 + (e - 8));
  int n = nt * 16 + (lane & 15);
  Wp[idx] = (k < K && n < Nc) ? f2bf(W[(size_t)k * Nc + n]) : f2bf(0.0f);
}

// ---------------- WMMA GEMM: C[M,Nc] = act(Abf[M,Kpad] @ W + bias) ----------------
// 4 waves/block, one 16x16 C tile per wave; B fragments staged in LDS (shared by all waves);
// A fragments are two guard-free global_load_b128 per lane per 32-K step.
__global__ void k_gemm_wmma(const bf16_t* __restrict__ Abf, int Kpad,
                            const bf16_t* __restrict__ Wp,
                            const float* __restrict__ bias,
                            float* __restrict__ C, int ldc,
                            int M, int Nc, int numKT, int numNT, int act) {
  __shared__ bf16_t Bs[8 * 512];            // up to 8 K-tiles of packed B (8 KB)
  int lane = threadIdx.x & 31;
  int wv   = threadIdx.x >> 5;
  int nt   = blockIdx.y;

  // cooperative stage of all B K-tiles for this N-tile into LDS (32-byte chunks)
  for (int c = threadIdx.x; c < numKT * 32; c += 128) {
    *reinterpret_cast<v16bf*>(&Bs[c * 16]) =
        *reinterpret_cast<const v16bf*>(&Wp[((size_t)((c >> 5) * numNT + nt) * 32 + (c & 31)) * 16]);
  }
  __syncthreads();

  int tileM = (blockIdx.x * 4 + wv) * 16;
  if (tileM >= M) return;                   // M is always a multiple of 16 here
  int hi = lane >> 4;

  union { v8f v; float f[8]; } acc;
#pragma unroll
  for (int r = 0; r < 8; ++r) acc.f[r] = 0.0f;

  const bf16_t* arow = Abf + (size_t)(tileM + (lane & 15)) * Kpad;
  for (int kt = 0; kt < numKT; ++kt) {
    int k0 = kt * 32;
    // A fragment: ISA 16-bit A 16x32 layout (lanes 0-15: K 0-7 & 16-23; lanes 16-31: K 8-15 & 24-31)
    union { v16bf v; v8bf h[2]; } af;
    af.h[0] = *reinterpret_cast<const v8bf*>(arow + k0 + hi * 8);
    af.h[1] = *reinterpret_cast<const v8bf*>(arow + k0 + 16 + hi * 8);
    // B fragment from LDS: ds_load_b128 pair
    const v16bf bfv = *reinterpret_cast<const v16bf*>(&Bs[(kt * 32 + lane) * 16]);

    acc.v = __builtin_amdgcn_wmma_f32_16x16x32_bf16(
        false, af.v, false, bfv, (short)0, acc.v, false, false);

    if (kt + 1 < numKT)
      __builtin_prefetch(arow + k0 + 32, 0, 1);   // global_prefetch_b8 for next A chunk
  }

  int col = nt * 16 + (lane & 15);
  float bv = (bias != nullptr && col < Nc) ? bias[col] : 0.0f;
#pragma unroll
  for (int r = 0; r < 8; ++r) {
    int row = tileM + hi * 8 + r;           // C layout: VGPR r -> M=r (lanes 0-15) / M=8+r (16-31)
    if (col < Nc) {
      float v = acc.f[r] + bv;
      if (act == ACT_RELU) v = fmaxf(v, 0.0f);
      else if (act == ACT_LEAKY01) v = leakyf(v, 0.01f);
      C[(size_t)row * ldc + col] = v;
    }
  }
}

// ---------------- segment_sum of concat(h_n[src], h_e) into agg[dst] ----------------
__global__ void k_scatter_agg(const float* __restrict__ hn, const float* __restrict__ he,
                              const int* __restrict__ src, const int* __restrict__ dst,
                              float* __restrict__ agg) {
  int e = blockIdx.x;
  int c = threadIdx.x;
  if (c >= FD) return;
  int d = dst[e];
  float v = (c < NFD) ? hn[(size_t)src[e] * NFD + c]
                      : he[(size_t)e * EFD + (c - NFD)];
  atomicAdd(&agg[(size_t)d * FD + c], v);
}

// ---------------- el/er = einsum(z, attn) ----------------
__global__ void k_attn_proj(const float* __restrict__ z,
                            const float* __restrict__ attnL, const float* __restrict__ attnR,
                            float* __restrict__ el, float* __restrict__ er) {
  int idx = blockIdx.x * blockDim.x + threadIdx.x;
  if (idx >= NNODE * NHD) return;
  int n = idx >> 2, h = idx & 3;
  const float* zr = z + (size_t)n * ZD + h * EMBD;
  const float* al = attnL + h * EMBD;
  const float* ar = attnR + h * EMBD;
  float sl = 0.0f, sr = 0.0f;
#pragma unroll 8
  for (int d = 0; d < EMBD; ++d) { sl += zr[d] * al[d]; sr += zr[d] * ar[d]; }
  el[idx] = sl; er[idx] = sr;
}

// ---------------- edge softmax pass 1: segment max ----------------
__global__ void k_edge_max(const float* __restrict__ el, const float* __restrict__ er,
                           const int* __restrict__ src, const int* __restrict__ dst,
                           float* __restrict__ mbuf) {
  int idx = blockIdx.x * blockDim.x + threadIdx.x;
  if (idx >= NEDGE * NHD) return;
  int e = idx >> 2, h = idx & 3;
  float v = leakyf(el[src[e] * NHD + h] + er[dst[e] * NHD + h], 0.2f);
  atomicMaxF(&mbuf[dst[e] * NHD + h], v);
}

// ---------------- edge softmax pass 2: exp + segment sum ----------------
__global__ void k_edge_exp(const float* __restrict__ el, const float* __restrict__ er,
                           const int* __restrict__ src, const int* __restrict__ dst,
                           const float* __restrict__ mbuf,
                           float* __restrict__ ex, float* __restrict__ den) {
  int idx = blockIdx.x * blockDim.x + threadIdx.x;
  if (idx >= NEDGE * NHD) return;
  int e = idx >> 2, h = idx & 3;
  int d = dst[e];
  float v = leakyf(el[src[e] * NHD + h] + er[d * NHD + h], 0.2f);
  float xv = __expf(v - mbuf[d * NHD + h]);
  ex[idx] = xv;
  atomicAdd(&den[d * NHD + h], xv);
}

// ---------------- attention aggregation: rst[dst] += a * z[src] ----------------
__global__ void k_edge_agg(const float* __restrict__ z, const float* __restrict__ ex,
                           const float* __restrict__ den,
                           const int* __restrict__ src, const int* __restrict__ dst,
                           float* __restrict__ rst) {
  int e = blockIdx.x;
  int c = threadIdx.x;            // 0..255, h = c>>6
  int h = c >> 6;
  int s = src[e], d = dst[e];
  float coef = ex[e * NHD + h] / den[d * NHD + h];
  atomicAdd(&rst[(size_t)d * ZD + c], coef * z[(size_t)s * ZD + c]);
}

// ---------------- rst = leaky(rst + gat_b, 0.01) ----------------
__global__ void k_rst_bias(float* __restrict__ rst, const float* __restrict__ gatB) {
  long long idx = (long long)blockIdx.x * blockDim.x + threadIdx.x;
  if (idx >= (long long)NNODE * ZD) return;
  int c = (int)(idx & 255);
  rst[idx] = leakyf(rst[idx] + gatB[c], 0.01f);
}

// ---------------- WeightedSumAndMax readout: one wave32 per node ----------------
__global__ void k_readout(const float* __restrict__ x,
                          const float* __restrict__ wsumW, const float* __restrict__ wsumB,
                          const int* __restrict__ gid,
                          float* __restrict__ hsum, float* __restrict__ hmax) {
  int node = blockIdx.x * 8 + (threadIdx.x >> 5);
  int lane = threadIdx.x & 31;
  if (node >= NNODE) return;
  float x0 = x[(size_t)node * EMBD + lane];
  float x1 = x[(size_t)node * EMBD + 32 + lane];
  float part = x0 * wsumW[lane] + x1 * wsumW[32 + lane];
#pragma unroll
  for (int off = 16; off > 0; off >>= 1) part += __shfl_xor(part, off, 32);
  float w = 1.0f / (1.0f + __expf(-(part + wsumB[0])));
  int g = gid[node];
  atomicAdd(&hsum[(size_t)g * EMBD + lane],      w * x0);
  atomicAdd(&hsum[(size_t)g * EMBD + 32 + lane], w * x1);
  atomicMaxF(&hmax[(size_t)g * EMBD + lane],      x0);
  atomicMaxF(&hmax[(size_t)g * EMBD + 32 + lane], x1);
}

// ---------------- hg = [hsum, where(finite, hmax, 0)] ----------------
__global__ void k_concat_hg(const float* __restrict__ hsum, const float* __restrict__ hmax,
                            float* __restrict__ hg) {
  int idx = blockIdx.x * blockDim.x + threadIdx.x;
  if (idx >= NGRF * 2 * EMBD) return;
  int b = idx >> 7, c = idx & 127;
  float v;
  if (c < EMBD) v = hsum[(size_t)b * EMBD + c];
  else { float m = hmax[(size_t)b * EMBD + (c - EMBD)]; v = (m < -3.0e38f) ? 0.0f : m; }
  hg[idx] = v;
}

// ---------------- host ----------------
static inline long long cdiv(long long a, long long b) { return (a + b - 1) / b; }

extern "C" void kernel_launch(void* const* d_in, const int* in_sizes, int n_in,
                              void* d_out, int out_size, void* d_ws, size_t ws_size,
                              hipStream_t stream) {
  (void)in_sizes; (void)n_in; (void)out_size; (void)ws_size;
  const float* const* F = (const float* const*)d_in;

  // ---- input index map (depth-first insertion-order flatten of setup_inputs) ----
  // per-branch params (16 leaves): 0 Wn 1 bn 2 We 3 be 4 Wc 5 bc 6 gatW 7 attnL
  //                                8 attnR 9 gatB 10 W1 11 b1 12 Ww 13 bw 14 W2 15 b2
  const int PB_A = 0, PB_B = 16;
  const float* out1W = F[32]; const float* out1b = F[33];
  const float* out2W = F[34]; const float* out2b = F[35];
  const int DATA0 = 36;  // node_a, edge_a, src_a, dst_a, gid_a, node_b, edge_b, src_b, dst_b, gid_b

  // ---- workspace layout (floats) ----
  float* W = (float*)d_ws;
  size_t o = 0;
  auto take = [&](size_t nf) { size_t r = o; o += (nf + 15) & ~(size_t)15; return r; };
  size_t o_hn   = take((size_t)NNODE * NFD);
  size_t o_he   = take((size_t)NEDGE * EFD);
  size_t o_agg  = take((size_t)NNODE * FD);
  size_t o_h    = take((size_t)NNODE * FD);
  size_t o_z    = take((size_t)NNODE * ZD);
  size_t o_el   = take((size_t)NNODE * NHD);
  size_t o_er   = take((size_t)NNODE * NHD);
  size_t o_m    = take((size_t)NNODE * NHD);
  size_t o_den  = take((size_t)NNODE * NHD);
  size_t o_ex   = take((size_t)NEDGE * NHD);
  size_t o_rst  = take((size_t)NNODE * ZD);
  size_t o_x    = take((size_t)NNODE * EMBD);
  size_t o_hsum = take((size_t)NGRF * EMBD);
  size_t o_hmax = take((size_t)NGRF * EMBD);
  size_t o_hg   = take((size_t)NGRF * 2 * EMBD);
  size_t o_zcat = take((size_t)NGRF * 2 * EMBD);
  size_t o_z1   = take((size_t)NGRF * EMBD);
  size_t o_wp   = take((size_t)40960);         // 81920 bf16 for packed B (max 8*16*512)
  size_t o_abf  = take((size_t)6500000);       // 13M bf16 for converted A (max 400000*32 / 50000*256)
  bf16_t* Wp  = (bf16_t*)(W + o_wp);
  bf16_t* Abf = (bf16_t*)(W + o_abf);

  auto fill = [&](float* p, long long n, float v) {
    k_fill<<<(unsigned)cdiv(n, 256), 256, 0, stream>>>(p, n, v);
  };
  auto gemm = [&](const float* A, const float* Wf, const float* bias, float* C, int ldc,
                  long long M, int K, int Nc, int act) {
    int numKT = (K + 31) / 32, numNT = (Nc + 15) / 16;
    int Kpad  = numKT * 32;
    k_cvt_abf<<<(unsigned)cdiv(M * Kpad, 256), 256, 0, stream>>>(A, Abf, M, K, Kpad);
    int total = numKT * numNT * 512;
    k_pack_w<<<(unsigned)cdiv(total, 256), 256, 0, stream>>>(Wf, Wp, K, Nc, numNT, total);
    dim3 grid((unsigned)cdiv(M, 64), (unsigned)numNT);
    k_gemm_wmma<<<grid, 128, 0, stream>>>(Abf, Kpad, Wp, bias, C, ldc, (int)M, Nc,
                                          numKT, numNT, act);
  };

  auto run_branch = [&](int pb, int db, int col_off) {
    const float* Wn = F[pb + 0];  const float* bn = F[pb + 1];
    const float* We = F[pb + 2];  const float* be = F[pb + 3];
    const float* Wc = F[pb + 4];  const float* bc = F[pb + 5];
    const float* gatW = F[pb + 6];
    const float* aL = F[pb + 7];  const float* aR = F[pb + 8];
    const float* gatB = F[pb + 9];
    const float* W1 = F[pb + 10]; const float* b1 = F[pb + 11];
    const float* Ww = F[pb + 12]; const float* bw = F[pb + 13];
    const float* W2 = F[pb + 14]; const float* b2 = F[pb + 15];
    const float* nf = F[db + 0];  const float* ef = F[db + 1];
    const int* src = (const int*)d_in[db + 2];
    const int* dst = (const int*)d_in[db + 3];
    const int* gid = (const int*)d_in[db + 4];

    fill(W + o_agg,  (long long)NNODE * FD, 0.0f);
    fill(W + o_m,    (long long)NNODE * NHD, -INFINITY);
    fill(W + o_den,  (long long)NNODE * NHD, 0.0f);
    fill(W + o_rst,  (long long)NNODE * ZD, 0.0f);
    fill(W + o_hsum, (long long)NGRF * EMBD, 0.0f);
    fill(W + o_hmax, (long long)NGRF * EMBD, -INFINITY);

    gemm(nf, Wn, bn, W + o_hn, NFD, NNODE, NFD, NFD, ACT_NONE);           // node proj
    gemm(ef, We, be, W + o_he, EFD, NEDGE, EFD, EFD, ACT_NONE);           // edge proj
    k_scatter_agg<<<NEDGE, 160, 0, stream>>>(W + o_hn, W + o_he, src, dst, W + o_agg);
    gemm(W + o_agg, Wc, bc, W + o_h, FD, NNODE, FD, FD, ACT_RELU);        // EdgeGraphConv
    gemm(W + o_h, gatW, nullptr, W + o_z, ZD, NNODE, FD, ZD, ACT_NONE);   // GAT projection
    k_attn_proj<<<(unsigned)cdiv((long long)NNODE * NHD, 256), 256, 0, stream>>>(
        W + o_z, aL, aR, W + o_el, W + o_er);
    k_edge_max<<<(unsigned)cdiv((long long)NEDGE * NHD, 256), 256, 0, stream>>>(
        W + o_el, W + o_er, src, dst, W + o_m);
    k_edge_exp<<<(unsigned)cdiv((long long)NEDGE * NHD, 256), 256, 0, stream>>>(
        W + o_el, W + o_er, src, dst, W + o_m, W + o_ex, W + o_den);
    k_edge_agg<<<NEDGE, 256, 0, stream>>>(W + o_z, W + o_ex, W + o_den, src, dst, W + o_rst);
    k_rst_bias<<<(unsigned)cdiv((long long)NNODE * ZD, 256), 256, 0, stream>>>(W + o_rst, gatB);
    gemm(W + o_rst, W1, b1, W + o_x, EMBD, NNODE, ZD, EMBD, ACT_NONE);    // lin1
    k_readout<<<(unsigned)cdiv(NNODE, 8), 256, 0, stream>>>(
        W + o_x, Ww, bw, gid, W + o_hsum, W + o_hmax);
    k_concat_hg<<<(unsigned)cdiv((long long)NGRF * 2 * EMBD, 256), 256, 0, stream>>>(
        W + o_hsum, W + o_hmax, W + o_hg);
    gemm(W + o_hg, W2, b2, W + o_zcat + col_off, 2 * EMBD, NGRF, 2 * EMBD, EMBD, ACT_NONE);
  };

  run_branch(PB_A, DATA0 + 0, 0);      // xa -> zcat[:, 0:64]
  run_branch(PB_B, DATA0 + 5, EMBD);   // xb -> zcat[:, 64:128]

  // head: z = leaky(zcat @ out1 + b, 0.01); out = z @ out2 + b
  gemm(W + o_zcat, out1W, out1b, W + o_z1, EMBD, NGRF, 2 * EMBD, EMBD, ACT_LEAKY01);
  gemm(W + o_z1, out2W, out2b, (float*)d_out, 2, NGRF, EMBD, 2, ACT_NONE);
}
